// EdgePathNN_33663953666603
// MI455X (gfx1250) — matrix operand order repaired and checked by reading, not compiled, not run on hardware
//
#include <hip/hip_runtime.h>
#include <hip/hip_bf16.h>

#define HD 64
#define NCL 10

typedef __attribute__((ext_vector_type(16))) __bf16 v16bf;
typedef __attribute__((ext_vector_type(8)))  float  v8f;
typedef __attribute__((ext_vector_type(4)))  unsigned int u32x4;

union FragU { u32x4 q[2]; v16bf v; };

// A fragment, 16x32 bf16 (MxK): lanes 0-15 hold M=lane with K=0..7,16..23;
// lanes 16-31 hold M=lane-16 with K=8..15,24..31 (ISA 7.12.2 layout).
__device__ __forceinline__ v16bf load_a_frag(const __bf16* base, int row_stride, int lane) {
  int m  = lane & 15;
  int k0 = (lane >> 4) ? 8 : 0;
  FragU f;
  f.q[0] = *reinterpret_cast<const u32x4*>(base + m * row_stride + k0);
  f.q[1] = *reinterpret_cast<const u32x4*>(base + m * row_stride + k0 + 16);
  return f.v;
}

// B fragment, 32x16 bf16 (KxN): lanes 0-15 hold N=lane with K=0..7,16..23;
// lanes 16-31 hold N=lane-16 with K=8..15,24..31.
__device__ __forceinline__ v16bf load_b_frag(const __bf16* base, int row_stride, int lane) {
  int n  = lane & 15;
  int k0 = (lane >> 4) ? 8 : 0;
  v16bf v;
#pragma unroll
  for (int s = 0; s < 8; ++s) {
    v[s]     = base[(k0 + s) * row_stride + n];
    v[s + 8] = base[(k0 + 16 + s) * row_stride + n];
  }
  return v;
}

__device__ __forceinline__ v8f wmma_bf16(v16bf a, v16bf b, v8f c) {
  return __builtin_amdgcn_wmma_f32_16x16x32_bf16(false, a, false, b, (short)0, c, false, false);
}

__device__ __forceinline__ float sigf(float x) { return 1.0f / (1.0f + __expf(-x)); }

// Async memory->LDS copy, 16B per lane, tracked by ASYNCcnt (CDNA5).
__device__ __forceinline__ void async_load_b128(unsigned lds_addr, unsigned gl_off,
                                                unsigned long long gbase) {
  asm volatile("global_load_async_to_lds_b128 %0, %1, %2"
               :: "v"(lds_addr), "v"(gl_off), "s"(gbase)
               : "memory");
}
__device__ __forceinline__ void wait_async0() {
  asm volatile("s_wait_asynccnt 0x0" ::: "memory");
}

// ---------------------------------------------------------------- utilities
__global__ void epnn_zero_kernel(float* __restrict__ p, int n) {
  int t = blockIdx.x * blockDim.x + threadIdx.x;
  if (t < n) p[t] = 0.0f;
}

// out[r, t] = b[t] + sum_k x[r,k] * w[k,t]   (K <= 32, 64 out features)
__global__ void epnn_encode_kernel(const float* __restrict__ x, const float* __restrict__ w,
                                   const float* __restrict__ b, float* __restrict__ out, int K) {
  __shared__ float xr[32];
  int r = blockIdx.x, t = threadIdx.x;
  if (t < K) xr[t] = x[(size_t)r * K + t];
  __syncthreads();
  float acc = b[t];
  for (int k = 0; k < K; ++k) acc += xr[k] * w[k * HD + t];
  out[(size_t)r * HD + t] = acc;
}

// Build bf16 transposed LSTM weights: WxT[k][n]=w_ih[n][k] (128x256),
// WhT[k][n]=w_hh[n][k] (64x256), bias[n]=b_ih[n]+b_hh[n].
__global__ void epnn_wconv_kernel(const float* __restrict__ w_ih, const float* __restrict__ w_hh,
                                  const float* __restrict__ b_ih, const float* __restrict__ b_hh,
                                  __bf16* __restrict__ WxT, __bf16* __restrict__ WhT,
                                  float* __restrict__ bias) {
  int t = blockIdx.x * blockDim.x + threadIdx.x;
  if (t < 128 * 256) {
    int k = t >> 8, n = t & 255;
    WxT[t] = (__bf16)w_ih[n * 128 + k];
  }
  int t2 = t - 128 * 256;
  if (t2 >= 0 && t2 < 64 * 256) {
    int k = t2 >> 8, n = t2 & 255;
    WhT[t2] = (__bf16)w_hh[n * 64 + k];
  }
  int t3 = t2 - 64 * 256;
  if (t3 >= 0 && t3 < 256) bias[t3] = b_ih[t3] + b_hh[t3];
}

// ---------------------------------------------------- fused LSTM path conv
// Block = 128 threads (4 waves). Wave handles 16 paths. Weights live in LDS
// (staged with global_load_async_to_lds_b128). Per step: stage x_cat (bf16)
// -> 16 N-tiles of gates via WMMA -> recurrence in LDS -> final hidden
// scatter-added with global f32 atomics.
template <int L>
__global__ void epnn_conv_lstm_kernel(const float* __restrict__ xin, const float* __restrict__ ea,
                                      const int* __restrict__ paths, const int* __restrict__ ei,
                                      const __bf16* __restrict__ WxTg, /* WhT contiguous after */
                                      const float* __restrict__ biasg,
                                      float* __restrict__ out_acc, int P) {
  extern __shared__ char smem[];
  __bf16* sWx  = (__bf16*)smem;                 // 128*256 (WhT 64*256 follows contiguously)
  __bf16* sWh  = sWx + 128 * 256;
  float*  sBias = (float*)(sWh + 64 * 256);     // 256
  char*   pw   = (char*)(sBias + 256);
  const int PER_WAVE = (16 * 128 + 16 * 64) * 2 + (16 * 256 + 16 * 64) * 4; // 26624 B
  int wave = threadIdx.x >> 5, lane = threadIdx.x & 31;
  char* my = pw + wave * PER_WAVE;
  __bf16* xstage = (__bf16*)my;                // 16 x 128
  __bf16* hstage = xstage + 16 * 128;          // 16 x 64
  float*  gates  = (float*)(hstage + 16 * 64); // 16 x 256
  float*  cbuf   = gates + 16 * 256;           // 16 x 64

  // async DMA of both weight matrices (98304 B) straight into LDS
  {
    unsigned lds0 = (unsigned)(size_t)sWx;
    unsigned long long gb = (unsigned long long)(size_t)WxTg;
    const int CHUNKS = (128 * 256 + 64 * 256) * 2 / 16; // 6144 x 16B
    for (int i = threadIdx.x; i < CHUNKS; i += blockDim.x)
      async_load_b128(lds0 + i * 16, (unsigned)(i * 16), gb);
    for (int i = threadIdx.x; i < 256; i += blockDim.x) sBias[i] = biasg[i];
    wait_async0();
  }
  __syncthreads();

  int tileBase = (blockIdx.x * 4 + wave) * 16;
  bool fullTile = (tileBase + 16) <= P;

  for (int l = 0; l < L; ++l) {
    // stage x_cat[p, l, 0:128] as bf16
    for (int t = lane; t < 16 * 128; t += 32) {
      int p = t >> 7, f = t & 127;
      int gp = tileBase + p;
      int cp = gp < P ? gp : 0;
      float v;
      if (f < HD) {
        int node = paths[cp * L + l];
        v = xin[(size_t)node * HD + f];
      } else if (l == 0) {
        v = 0.0f;
      } else {
        int e = ei[cp * (L - 1) + (l - 1)];
        v = ea[(size_t)e * HD + (f - HD)];
      }
      xstage[t] = (__bf16)v;
    }
    if (l == 0) {
      for (int t = lane; t < 16 * 64; t += 32) cbuf[t] = 0.0f;
    }
    __syncthreads();

    // gates[p, 0:256] = x_cat @ WxT + (h @ WhT) + bias
    for (int nt = 0; nt < 16; ++nt) {
      float bv = sBias[nt * 16 + (lane & 15)];
      v8f acc = {bv, bv, bv, bv, bv, bv, bv, bv};
#pragma unroll
      for (int kt = 0; kt < 4; ++kt) {
        v16bf a = load_a_frag(xstage + kt * 32, 128, lane);
        v16bf b = load_b_frag(sWx + (kt * 32) * 256 + nt * 16, 256, lane);
        acc = wmma_bf16(a, b, acc);
      }
      if (l > 0) {
#pragma unroll
        for (int kt = 0; kt < 2; ++kt) {
          v16bf a = load_a_frag(hstage + kt * 32, 64, lane);
          v16bf b = load_b_frag(sWh + (kt * 32) * 256 + nt * 16, 256, lane);
          acc = wmma_bf16(a, b, acc);
        }
      }
      int n = nt * 16 + (lane & 15);
      int mb = (lane >> 4) ? 8 : 0;
#pragma unroll
      for (int r = 0; r < 8; ++r) gates[(mb + r) * 256 + n] = acc[r];
    }
    __syncthreads();

    // recurrence (torch gate order i,f,g,o)
    for (int t = lane; t < 16 * 64; t += 32) {
      int p = t >> 6, u = t & 63;
      float gi = gates[p * 256 + u];
      float gf = gates[p * 256 + 64 + u];
      float gg = gates[p * 256 + 128 + u];
      float go = gates[p * 256 + 192 + u];
      float c = sigf(gf) * cbuf[t] + sigf(gi) * tanhf(gg);
      float h = sigf(go) * tanhf(c);
      cbuf[t] = c;
      hstage[t] = (__bf16)h;
      if (l == L - 1) {
        int gp = tileBase + p;
        if (fullTile) {
          int dest = paths[gp * L + (L - 1)];
          atomicAdd(&out_acc[(size_t)dest * HD + u], h);
        } else if (gp < P) {
          int dest = paths[gp * L + (L - 1)];
          atomicAdd(&out_acc[(size_t)dest * HD + u], h);
        }
      }
    }
    __syncthreads();
  }
}

// ------------------------------------------------------- BN stats / BN+GEMM
__global__ void epnn_colstats_kernel(const float* __restrict__ h, float* __restrict__ stats,
                                     int n_rows) {
  int f = threadIdx.x & 63, rg = threadIdx.x >> 6;
  float s = 0.0f, s2 = 0.0f;
  for (int n = blockIdx.x * 4 + rg; n < n_rows; n += gridDim.x * 4) {
    float v = h[(size_t)n * HD + f];
    s += v; s2 += v * v;
  }
  __shared__ float ls[256], ls2[256];
  ls[threadIdx.x] = s; ls2[threadIdx.x] = s2;
  __syncthreads();
  if (rg == 0) {
    atomicAdd(&stats[f],      ls[f] + ls[64 + f] + ls[128 + f] + ls[192 + f]);
    atomicAdd(&stats[64 + f], ls2[f] + ls2[64 + f] + ls2[128 + f] + ls2[192 + f]);
  }
}

// out = (relu?)(BN(in)) @ W + wb ; accumulates column sum/sumsq of out.
__global__ void epnn_bn_gemm_kernel(const float* __restrict__ in, const float* __restrict__ instats,
                                    const float* __restrict__ bng, const float* __restrict__ bnb,
                                    int relu_in, const float* __restrict__ W,
                                    const float* __restrict__ wb, float* __restrict__ out,
                                    float* __restrict__ outstats, int n_rows, float inv_n) {
  __shared__ __bf16 sW[64 * 64];
  __shared__ float sScale[64], sShift[64];
  __shared__ __bf16 sA[4][16 * 64];
  int tid = threadIdx.x, wave = tid >> 5, lane = tid & 31;
  for (int i = tid; i < 64 * 64; i += blockDim.x) sW[i] = (__bf16)W[i];
  if (tid < 64) {
    float m = instats[tid] * inv_n;
    float var = instats[64 + tid] * inv_n - m * m;
    float rs = rsqrtf(var + 1e-5f);
    sScale[tid] = rs * bng[tid];
    sShift[tid] = bnb[tid] - m * rs * bng[tid];
  }
  __syncthreads();
  int rowBase = (blockIdx.x * 4 + wave) * 16;
  bool fullTile = (rowBase + 16) <= n_rows;
  for (int t = lane; t < 16 * 64; t += 32) {
    int p = t >> 6, f = t & 63;
    int r = rowBase + p;
    float v = 0.0f;
    if (r < n_rows) {
      v = in[(size_t)r * HD + f] * sScale[f] + sShift[f];
      if (relu_in && v < 0.0f) v = 0.0f;
    }
    sA[wave][t] = (__bf16)v;
  }
  __syncthreads();
  for (int nt = 0; nt < 4; ++nt) {
    int n = nt * 16 + (lane & 15);
    float bv = wb[n];
    v8f acc = {bv, bv, bv, bv, bv, bv, bv, bv};
#pragma unroll
    for (int kt = 0; kt < 2; ++kt) {
      v16bf a = load_a_frag(sA[wave] + kt * 32, 64, lane);
      v16bf b = load_b_frag(sW + (kt * 32) * 64 + nt * 16, 64, lane);
      acc = wmma_bf16(a, b, acc);
    }
    int mb = (lane >> 4) ? 8 : 0;
    float s = 0.0f, s2 = 0.0f;
    if (fullTile) {
      float* po = out + (size_t)(rowBase + mb) * HD + n;
#pragma unroll
      for (int r = 0; r < 8; ++r) {
        po[(size_t)r * HD] = acc[r];
        s += acc[r]; s2 += acc[r] * acc[r];
      }
    } else {
#pragma unroll
      for (int r = 0; r < 8; ++r) {
        int row = rowBase + mb + r;
        if (row < n_rows) {
          out[(size_t)row * HD + n] = acc[r];
          s += acc[r]; s2 += acc[r] * acc[r];
        }
      }
    }
    atomicAdd(&outstats[n], s);
    atomicAdd(&outstats[64 + n], s2);
  }
}

__global__ void epnn_bn_relu_kernel(const float* __restrict__ in, const float* __restrict__ stats,
                                    const float* __restrict__ g, const float* __restrict__ b,
                                    float* __restrict__ out, int n_rows, float inv_n) {
  int t = blockIdx.x * blockDim.x + threadIdx.x;
  if (t >= n_rows * HD) return;
  int f = t & 63;
  float m = stats[f] * inv_n;
  float var = stats[64 + f] * inv_n - m * m;
  float rs = rsqrtf(var + 1e-5f);
  float v = (in[t] - m) * rs * g[f] + b[f];
  out[t] = v > 0.0f ? v : 0.0f;
}

__global__ void epnn_axpby_kernel(const float* __restrict__ a, const float* __restrict__ b,
                                  float ca, float cb, float* __restrict__ out, int n) {
  int t = blockIdx.x * blockDim.x + threadIdx.x;
  if (t < n) out[t] = ca * a[t] + cb * b[t];
}

// --------------------------------------------------------------- attention
__global__ void epnn_coldot_kernel(const float* __restrict__ a, const float* __restrict__ b,
                                   float* __restrict__ out, int n_rows) {
  int f = threadIdx.x & 63, rg = threadIdx.x >> 6;
  float s = 0.0f;
  for (int n = blockIdx.x * 4 + rg; n < n_rows; n += gridDim.x * 4)
    s += a[(size_t)n * HD + f] * b[(size_t)n * HD + f];
  __shared__ float ls[256];
  ls[threadIdx.x] = s;
  __syncthreads();
  if (rg == 0) atomicAdd(&out[f], ls[f] + ls[64 + f] + ls[128 + f] + ls[192 + f]);
}

__global__ void epnn_att_finalize_kernel(const float* __restrict__ scores,
                                         const float* __restrict__ w_att,
                                         const float* __restrict__ b_att,
                                         float* __restrict__ wts) {
  __shared__ float red[64];
  __shared__ float sc[2];
  int t = threadIdx.x;
  for (int j = 0; j < 2; ++j) {
    float v = scores[j * 64 + t];
    red[t] = v; __syncthreads();
    for (int s = 32; s > 0; s >>= 1) { if (t < s) red[t] = fminf(red[t], red[t + s]); __syncthreads(); }
    float mn = red[0]; __syncthreads();
    red[t] = v; __syncthreads();
    for (int s = 32; s > 0; s >>= 1) { if (t < s) red[t] = fmaxf(red[t], red[t + s]); __syncthreads(); }
    float mx = red[0]; __syncthreads();
    float norm = (v - mn) / (mx - mn + 1e-6f);
    red[t] = norm * w_att[t]; __syncthreads();
    for (int s = 32; s > 0; s >>= 1) { if (t < s) red[t] += red[t + s]; __syncthreads(); }
    if (t == 0) sc[j] = red[0] + b_att[0];
    __syncthreads();
  }
  if (t == 0) {
    float m = fmaxf(sc[0], sc[1]);
    float e0 = __expf(sc[0] - m), e1 = __expf(sc[1] - m);
    wts[0] = e0 / (e0 + e1);
    wts[1] = e1 / (e0 + e1);
  }
}

__global__ void epnn_rep_pool_kernel(const float* __restrict__ W0, const float* __restrict__ W1,
                                     const float* __restrict__ W2, const float* __restrict__ wts,
                                     const int* __restrict__ batch, float* __restrict__ pooled,
                                     int n_rows) {
  int t = blockIdx.x * blockDim.x + threadIdx.x;
  if (t >= n_rows * HD) return;
  int nrow = t >> 6, f = t & 63;
  float rep = W0[t] + wts[0] * W1[t] + wts[1] * W2[t];
  atomicAdd(&pooled[(size_t)batch[nrow] * HD + f], rep);
}

__global__ void epnn_head_kernel(const float* __restrict__ pooled, const float* __restrict__ w1,
                                 const float* __restrict__ b1, const float* __restrict__ w2,
                                 const float* __restrict__ b2, float* __restrict__ out) {
  __shared__ float row[64], hid[64];
  int g = blockIdx.x, t = threadIdx.x;
  row[t] = pooled[(size_t)g * HD + t];
  __syncthreads();
  float acc = b1[t];
  for (int k = 0; k < 64; ++k) acc += row[k] * w1[k * HD + t];
  hid[t] = acc > 0.0f ? acc : 0.0f;
  __syncthreads();
  if (t < NCL) {
    float o = b2[t];
    for (int k = 0; k < 64; ++k) o += hid[k] * w2[k * NCL + t];
    out[(size_t)g * NCL + t] = o;
  }
}

// ------------------------------------------------------------------ launch
extern "C" void kernel_launch(void* const* d_in, const int* in_sizes, int n_in,
                              void* d_out, int out_size, void* d_ws, size_t ws_size,
                              hipStream_t stream) {
  const float* x         = (const float*)d_in[0];
  const float* edge_attr = (const float*)d_in[1];
  const int*   paths2    = (const int*)d_in[2];
  const int*   ei2       = (const int*)d_in[3];
  const int*   paths3    = (const int*)d_in[4];
  const int*   ei3       = (const int*)d_in[5];
  const int*   batch     = (const int*)d_in[6];
  const float* w_feat    = (const float*)d_in[7];
  const float* b_feat    = (const float*)d_in[8];
  const float* w_bond    = (const float*)d_in[9];
  const float* b_bond    = (const float*)d_in[10];
  const float* w_ih      = (const float*)d_in[11];
  const float* w_hh      = (const float*)d_in[12];
  const float* b_ih      = (const float*)d_in[13];
  const float* b_hh      = (const float*)d_in[14];
  const float* bn_g      = (const float*)d_in[15];
  const float* bn_b      = (const float*)d_in[16];
  const float* mlp_w1    = (const float*)d_in[17];
  const float* mlp_b1    = (const float*)d_in[18];
  const float* bn1_g     = (const float*)d_in[19];
  const float* bn1_b     = (const float*)d_in[20];
  const float* mlp_w2    = (const float*)d_in[21];
  const float* mlp_b2    = (const float*)d_in[22];
  const float* bn2_g     = (const float*)d_in[23];
  const float* bn2_b     = (const float*)d_in[24];
  const float* w_att     = (const float*)d_in[25];
  const float* b_att     = (const float*)d_in[26];
  const float* w_l1      = (const float*)d_in[27];
  const float* b_l1      = (const float*)d_in[28];
  const float* w_l2      = (const float*)d_in[29];
  const float* b_l2      = (const float*)d_in[30];
  (void)n_in; (void)ws_size;

  const int NF = 18, EF = 18;
  const int N  = in_sizes[0] / NF;
  const int E  = in_sizes[1] / EF;
  const int P2 = in_sizes[2] / 2;
  const int P3 = in_sizes[4] / 3;
  const int G  = out_size / NCL;
  const float inv_n = 1.0f / (float)N;
  float* out = (float*)d_out;

  // ---------------- workspace carve (float units) ----------------
  float* ws = (float*)d_ws;
  size_t off = 0;
  float* W0  = ws + off; off += (size_t)N * HD;
  float* EA  = ws + off; off += (size_t)E * HD;
  float* ACC = ws + off; off += (size_t)N * HD;
  float* Y1  = ws + off; off += (size_t)N * HD;
  float* Y2  = ws + off; off += (size_t)N * HD;
  float* W1  = ws + off; off += (size_t)N * HD;
  float* W2  = ws + off; off += (size_t)N * HD;
  float* WIN = ws + off; off += (size_t)N * HD;
  __bf16* WxT = (__bf16*)(ws + off); off += 16384; // 128*256 bf16
  __bf16* WhT = (__bf16*)(ws + off); off += 8192;  // 64*256 bf16, contiguous after WxT
  float* BIAS   = ws + off; off += 256;
  float* STA    = ws + off; off += 128;
  float* STB    = ws + off; off += 128;
  float* STC    = ws + off; off += 128;
  float* SCORES = ws + off; off += 128;
  float* WTS    = ws + off; off += 64;
  float* POOLED = ws + off; off += (size_t)G * HD;

  const int NW = (int)((size_t)N * HD);
  const int CONV_SMEM = 128 * 256 * 2 + 64 * 256 * 2 + 256 * 4 +
                        4 * ((16 * 128 + 16 * 64) * 2 + (16 * 256 + 16 * 64) * 4); // 205824 B

  // ---------------- encoders & weight prep ----------------
  epnn_encode_kernel<<<N, 64, 0, stream>>>(x, w_feat, b_feat, W0, NF);
  epnn_encode_kernel<<<E, 64, 0, stream>>>(edge_attr, w_bond, b_bond, EA, EF);
  epnn_wconv_kernel<<<(128 * 256 + 64 * 256 + 256 + 255) / 256, 256, 0, stream>>>(
      w_ih, w_hh, b_ih, b_hh, WxT, WhT, BIAS);

  // ---------------- conv 0 (L=2) ----------------
  epnn_zero_kernel<<<(NW + 255) / 256, 256, 0, stream>>>(ACC, NW);
  epnn_zero_kernel<<<1, 256, 0, stream>>>(STA, 128);
  epnn_zero_kernel<<<1, 256, 0, stream>>>(STB, 128);
  epnn_zero_kernel<<<1, 256, 0, stream>>>(STC, 128);
  epnn_conv_lstm_kernel<2><<<(P2 + 63) / 64, 128, CONV_SMEM, stream>>>(
      W0, EA, paths2, ei2, WxT, BIAS, ACC, P2);
  epnn_colstats_kernel<<<256, 256, 0, stream>>>(ACC, STA, N);
  epnn_bn_gemm_kernel<<<(N + 63) / 64, 128, 0, stream>>>(
      ACC, STA, bn_g + 0 * HD, bn_b + 0 * HD, 0, mlp_w1 + 0 * HD * HD, mlp_b1 + 0 * HD,
      Y1, STB, N, inv_n);
  epnn_bn_gemm_kernel<<<(N + 63) / 64, 128, 0, stream>>>(
      Y1, STB, bn1_g + 0 * HD, bn1_b + 0 * HD, 1, mlp_w2 + 0 * HD * HD, mlp_b2 + 0 * HD,
      Y2, STC, N, inv_n);
  epnn_bn_relu_kernel<<<(NW + 255) / 256, 256, 0, stream>>>(
      Y2, STC, bn2_g + 0 * HD, bn2_b + 0 * HD, W1, N, inv_n);

  // Win = 0.75*W0 - 0.25*W1
  epnn_axpby_kernel<<<(NW + 255) / 256, 256, 0, stream>>>(W0, W1, 0.75f, -0.25f, WIN, NW);

  // ---------------- conv 1 (L=3) ----------------
  epnn_zero_kernel<<<(NW + 255) / 256, 256, 0, stream>>>(ACC, NW);
  epnn_zero_kernel<<<1, 256, 0, stream>>>(STA, 128);
  epnn_zero_kernel<<<1, 256, 0, stream>>>(STB, 128);
  epnn_zero_kernel<<<1, 256, 0, stream>>>(STC, 128);
  epnn_conv_lstm_kernel<3><<<(P3 + 63) / 64, 128, CONV_SMEM, stream>>>(
      WIN, EA, paths3, ei3, WxT, BIAS, ACC, P3);
  epnn_colstats_kernel<<<256, 256, 0, stream>>>(ACC, STA, N);
  epnn_bn_gemm_kernel<<<(N + 63) / 64, 128, 0, stream>>>(
      ACC, STA, bn_g + 1 * HD, bn_b + 1 * HD, 0, mlp_w1 + 1 * HD * HD, mlp_b1 + 1 * HD,
      Y1, STB, N, inv_n);
  epnn_bn_gemm_kernel<<<(N + 63) / 64, 128, 0, stream>>>(
      Y1, STB, bn1_g + 1 * HD, bn1_b + 1 * HD, 1, mlp_w2 + 1 * HD * HD, mlp_b2 + 1 * HD,
      Y2, STC, N, inv_n);
  epnn_bn_relu_kernel<<<(NW + 255) / 256, 256, 0, stream>>>(
      Y2, STC, bn2_g + 1 * HD, bn2_b + 1 * HD, W2, N, inv_n);

  // ---------------- attention + pool + head ----------------
  epnn_zero_kernel<<<1, 256, 0, stream>>>(SCORES, 128);
  epnn_zero_kernel<<<((int)((size_t)G * HD) + 255) / 256, 256, 0, stream>>>(POOLED, G * HD);
  epnn_coldot_kernel<<<256, 256, 0, stream>>>(W0, W1, SCORES + 0, N);
  epnn_coldot_kernel<<<256, 256, 0, stream>>>(W0, W2, SCORES + 64, N);
  epnn_att_finalize_kernel<<<1, 64, 0, stream>>>(SCORES, w_att, b_att, WTS);
  epnn_rep_pool_kernel<<<(NW + 255) / 256, 256, 0, stream>>>(W0, W1, W2, WTS, batch, POOLED, N);
  epnn_head_kernel<<<G, 64, 0, stream>>>(POOLED, w_l1, b_l1, w_l2, b_l2, out);
  (void)WhT;
}
